// SelfAttentionHead_37967510896887
// MI455X (gfx1250) — matrix operand order repaired and checked by reading, not compiled, not run on hardware
//
#include <hip/hip_runtime.h>
#include <hip/hip_bf16.h>

// ---- trivially-copyable vector types (avoid HIP_vector_type in unions) ----
typedef _Float16     v16h __attribute__((ext_vector_type(16)));
typedef float        v8f  __attribute__((ext_vector_type(8)));
typedef float        v4f  __attribute__((ext_vector_type(4)));
typedef unsigned int v8u  __attribute__((ext_vector_type(8)));
typedef unsigned int v4u  __attribute__((ext_vector_type(4)));

union U16H { v4u u4[2]; v8u u8; v16h h; };   // 32B: one WMMA f16 A/B operand
union PK8  { v4u u4; _Float16 h[8]; };        // 16B: 8 packed halves

static __device__ __forceinline__ v8f wmma_f16(v16h a, v16h b, v8f c) {
  return __builtin_amdgcn_wmma_f32_16x16x32_f16(false, a, false, b, (short)0, c,
                                                false, false);
}
static __device__ __forceinline__ v8f vzero8() {
  v8f z = {0.f, 0.f, 0.f, 0.f, 0.f, 0.f, 0.f, 0.f};
  return z;
}

// dims
#define BB 8
#define TT 2048
#define CC 1024
#define HH 64

// ---------------------------------------------------------------------------
// Kernel 0: WT[t][n][k] = (f16) W_t[k][n]   (t: 0=Wq,1=Wk,2=Wv) — tiny, 384KB
// ---------------------------------------------------------------------------
__global__ void __launch_bounds__(256)
wt_build(const float* __restrict__ Wq, const float* __restrict__ Wk,
         const float* __restrict__ Wv, _Float16* __restrict__ WT) {
  int e = blockIdx.x * 256 + threadIdx.x;      // exactly 3*64*1024 threads
  int t = e >> 16;
  int rem = e & 65535;
  int n = rem & 63;
  int k = rem >> 6;
  const float* W = (t == 0) ? Wq : ((t == 1) ? Wk : Wv);
  WT[t * 65536 + n * 1024 + k] = (_Float16)W[k * 64 + n];
}

// ---------------------------------------------------------------------------
// Kernel 1: projections. One wave = one 16x64 tile of (X @ W).
// tensor 0: queries@Wq -> q16[t][h]   tensor 1: keys@Wk -> k16[t][h]
// tensor 2: values@Wv -> vT[h][t]  (transposed for the PV B-operand)
// Memory-bound: 192MB of f32 activations @23.3TB/s ~= 8.3us floor.
// ---------------------------------------------------------------------------
__global__ void __launch_bounds__(256)
proj_kernel(const float* __restrict__ Xq, const float* __restrict__ Xk,
            const float* __restrict__ Xv, const _Float16* __restrict__ WT,
            _Float16* __restrict__ q16, _Float16* __restrict__ k16,
            _Float16* __restrict__ vT) {
  const int lane = threadIdx.x & 31;
  const int wave = blockIdx.x * (blockDim.x >> 5) + (threadIdx.x >> 5);
  const int tensor = wave >> 10;   // 0..2
  const int mtile  = wave & 1023;  // 0..1023  (16 rows each over B*T)
  const float* X = (tensor == 0) ? Xq : ((tensor == 1) ? Xk : Xv);
  const _Float16* Wt = WT + tensor * 65536;

  const int g   = lane >> 4;     // half-group
  const int col = lane & 15;
  const int m   = mtile * 16 + col;   // A-operand row (lane&15 for all lanes)
  const int kbA = g * 8;              // A k-chunk base within 32-step
  const int kbB = g * 16;             // unused; B reads the same kbA chunks as A

  v8f acc[4];
#pragma unroll
  for (int n = 0; n < 4; ++n) acc[n] = vzero8();

  for (int ks = 0; ks < 32; ++ks) {
    const int k0 = ks * 32;
    // A: 16 halves per lane = f32 chunks [kb,kb+8) and [kb+16,kb+24)
    const float* ap = X + (long)m * CC + k0 + kbA;
    v4f f0 = *(const v4f*)(ap);
    v4f f1 = *(const v4f*)(ap + 4);
    v4f f2 = *(const v4f*)(ap + 16);
    v4f f3 = *(const v4f*)(ap + 20);
    v16h a;
#pragma unroll
    for (int i = 0; i < 4; ++i) {
      a[i]      = (_Float16)f0[i];
      a[4 + i]  = (_Float16)f1[i];
      a[8 + i]  = (_Float16)f2[i];
      a[12 + i] = (_Float16)f3[i];
    }
#pragma unroll
    for (int n = 0; n < 4; ++n) {
      U16H b;
      b.u4[0] = *(const v4u*)(Wt + (n * 16 + col) * 1024 + k0 + kbA);
      b.u4[1] = *(const v4u*)(Wt + (n * 16 + col) * 1024 + k0 + kbA + 16);
      acc[n] = wmma_f16(a, b.h, acc[n]);
    }
  }

  // epilogue: D layout: lane, VGPR r -> row r + g*8, col lane&15
  const int b  = mtile >> 7;                     // batch
  const int tb = (mtile & 127) * 16 + g * 8;     // t base for this half-group
  if (tensor < 2) {
    _Float16* dst = (tensor == 0) ? q16 : k16;
#pragma unroll
    for (int n = 0; n < 4; ++n)
#pragma unroll
      for (int r = 0; r < 8; ++r)
        dst[((b * TT + tb + r) * HH) + n * 16 + col] = (_Float16)acc[n][r];
  } else {
    // vT[b][h][t]: consecutive VGPRs are consecutive t -> packed 16B store
#pragma unroll
    for (int n = 0; n < 4; ++n) {
      PK8 pk;
#pragma unroll
      for (int r = 0; r < 8; ++r) pk.h[r] = (_Float16)acc[n][r];
      *(v4u*)(vT + ((long)(b * HH + n * 16 + col)) * TT + tb) = pk.u4;
    }
  }
}

// ---------------------------------------------------------------------------
// Kernel 2: flash attention, templated on split-S.
//   FINAL=true,  STRIDE=1: one wave per (b, q-tile); divides and writes out.
//   FINAL=false, STRIDE=4: 4 waves per (b, q-tile), seg takes chunks
//                          c ≡ seg (mod 4); writes partial (m,l,O) to ws.
// Split-S raises wave count 1024 -> 4096 (~4/SIMD) so L2/softmax/LDS latency
// is hidden, and cuts the i=127 critical path 64 -> 16 chunk iterations.
// ---------------------------------------------------------------------------
template <int STRIDE, bool FINAL>
__global__ void __launch_bounds__(128)
attn_main(const _Float16* __restrict__ q16, const _Float16* __restrict__ k16,
          const _Float16* __restrict__ vT, float* __restrict__ dst,
          float* __restrict__ mbuf, float* __restrict__ lbuf) {
  __shared__ __align__(16) _Float16 Pbuf[4][16 * 32];

  const int lane = threadIdx.x & 31;
  const int wv   = threadIdx.x >> 5;
  const int wave = blockIdx.x * 4 + wv;
  int b, i, seg;
  if (FINAL) {
    b = wave >> 7; i = wave & 127; seg = 0;          // 1024 waves
  } else {
    b = wave >> 9; i = (wave & 511) >> 2; seg = wave & 3;  // 4096 waves
  }
  const int g   = lane >> 4;
  const int col = lane & 15;
  const int kbA = g * 8;      // A-layout k-chunk base
  const int hbB = g * 16;     // B-layout k-run base
  _Float16* P = Pbuf[wv];
  const float NEG_INF = -__builtin_inff();

  // A_Q: rows t = 16i + (lane&15), two k-steps over h=64
  const _Float16* qp = q16 + (long)(b * TT + i * 16 + col) * HH;
  U16H aq0, aq1;
  aq0.u4[0] = *(const v4u*)(qp + hbB);
  aq0.u4[1] = *(const v4u*)(qp + hbB + 8);
  aq1.u4[0] = *(const v4u*)(qp + hbB + 32);
  aq1.u4[1] = *(const v4u*)(qp + hbB + 40);

  float mrun[8], lrun[8];
  v8f o[4];
#pragma unroll
  for (int r = 0; r < 8; ++r) { mrun[r] = NEG_INF; lrun[r] = 0.f; }
#pragma unroll
  for (int n = 0; n < 4; ++n) o[n] = vzero8();

  const int cmax = i >> 1;
  for (int c = seg; c <= cmax; c += STRIDE) {
    const int sb = c * 32;
    // B_K: two 16-col s tiles, two k-steps each (h runs of 16, contiguous)
    const _Float16* kp0 = k16 + (long)(b * TT + sb + col) * HH;
    const _Float16* kp1 = kp0 + 16 * HH;
    U16H bk00, bk01, bk10, bk11;
    bk00.u8 = *(const v8u*)(kp0 + hbB);
    bk01.u8 = *(const v8u*)(kp0 + 32 + hbB);
    bk10.u8 = *(const v8u*)(kp1 + hbB);
    bk11.u8 = *(const v8u*)(kp1 + 32 + hbB);

    v8f c0 = wmma_f16(aq0.h, bk00.h, vzero8());
    c0     = wmma_f16(aq1.h, bk01.h, c0);
    v8f c1 = wmma_f16(aq0.h, bk10.h, vzero8());
    c1     = wmma_f16(aq1.h, bk11.h, c1);

    // scale by C**-0.5 = 1/32, causal mask only in diagonal chunk
#pragma unroll
    for (int r = 0; r < 8; ++r) { c0[r] *= 0.03125f; c1[r] *= 0.03125f; }
    if (c == cmax) {
#pragma unroll
      for (int r = 0; r < 8; ++r) {
        const int t_abs = i * 16 + r + g * 8;
        if (sb + col > t_abs)      c0[r] = NEG_INF;
        if (sb + 16 + col > t_abs) c1[r] = NEG_INF;
      }
    }

    // online softmax: row r+g*8 lives in VGPR r across the 16-lane half-group
    float p0[8], p1[8], alpha[8];
#pragma unroll
    for (int r = 0; r < 8; ++r) {
      float rm = fmaxf(c0[r], c1[r]);
#pragma unroll
      for (int off = 1; off < 16; off <<= 1)
        rm = fmaxf(rm, __shfl_xor(rm, off, 32));
      const float mn = fmaxf(mrun[r], rm);
      alpha[r] = __expf(mrun[r] - mn);
      p0[r] = __expf(c0[r] - mn);
      p1[r] = __expf(c1[r] - mn);
      float rs = p0[r] + p1[r];
#pragma unroll
      for (int off = 1; off < 16; off <<= 1) rs += __shfl_xor(rs, off, 32);
      lrun[r] = lrun[r] * alpha[r] + rs;
      mrun[r] = mn;
    }
#pragma unroll
    for (int n = 0; n < 4; ++n)
#pragma unroll
      for (int r = 0; r < 8; ++r) o[n][r] *= alpha[r];

    // P: D-layout -> A-layout via wave-private LDS (HW keeps a wave's DS ops
    // in order; asm barriers stop compiler reordering across the staging)
#pragma unroll
    for (int r = 0; r < 8; ++r) {
      P[(r + g * 8) * 32 + col]      = (_Float16)p0[r];
      P[(r + g * 8) * 32 + 16 + col] = (_Float16)p1[r];
    }
    asm volatile("s_wait_dscnt 0" ::: "memory");
    U16H ap;
    ap.u4[0] = *(const v4u*)(P + col * 32 + hbB);
    ap.u4[1] = *(const v4u*)(P + col * 32 + hbB + 8);
    asm volatile("" ::: "memory");

    // O += P @ V  (B from vT[h][s]: per-lane contiguous s-run of 16)
#pragma unroll
    for (int n = 0; n < 4; ++n) {
      U16H bv;
      bv.u8 = *(const v8u*)(vT + (long)(b * HH + n * 16 + col) * TT + sb + hbB);
      o[n] = wmma_f16(ap.h, bv.h, o[n]);
    }
  }

  if (FINAL) {
    // divide by softmax sum, scatter f32 to out[b][t][h]
#pragma unroll
    for (int r = 0; r < 8; ++r) {
      const float inv = 1.0f / lrun[r];
      const int t = i * 16 + r + g * 8;
#pragma unroll
      for (int n = 0; n < 4; ++n)
        dst[(long)(b * TT + t) * HH + n * 16 + col] = o[n][r] * inv;
    }
  } else {
    // partial: dst = Obuf[(tile*4+seg)][row][h], plus per-row (m,l)
    const long pb = ((long)(b * 128 + i) * 4 + seg);
    float* Op = dst + pb * 1024;
#pragma unroll
    for (int r = 0; r < 8; ++r) {
      const int row = r + g * 8;
#pragma unroll
      for (int n = 0; n < 4; ++n)
        Op[row * 64 + n * 16 + col] = o[n][r];
    }
    if (col == 0) {   // lanes 0 and 16 own their half-group's rows (after WMMAs)
#pragma unroll
      for (int r = 0; r < 8; ++r) {
        mbuf[pb * 16 + r + g * 8] = mrun[r];
        lbuf[pb * 16 + r + g * 8] = lrun[r];
      }
    }
  }
}

// ---------------------------------------------------------------------------
// Kernel 3: merge the 4 split-S partials per (b, q-tile).
// out = sum_s e^{m_s-M} O_s / sum_s e^{m_s-M} l_s   (M = max_s m_s)
// ---------------------------------------------------------------------------
__global__ void __launch_bounds__(256)
attn_combine(const float* __restrict__ Obuf, const float* __restrict__ mbuf,
             const float* __restrict__ lbuf, float* __restrict__ out) {
  const int tile = blockIdx.x;          // b*128 + i
  const int b = tile >> 7, i = tile & 127;
#pragma unroll
  for (int k = 0; k < 4; ++k) {
    const int e = threadIdx.x + k * 256;   // row*64 + h, 0..1023
    const int row = e >> 6;
    float m[4], l[4];
#pragma unroll
    for (int s = 0; s < 4; ++s) {
      m[s] = mbuf[(tile * 4 + s) * 16 + row];
      l[s] = lbuf[(tile * 4 + s) * 16 + row];
    }
    float M = fmaxf(fmaxf(m[0], m[1]), fmaxf(m[2], m[3]));
    float L = 0.f, acc = 0.f;
#pragma unroll
    for (int s = 0; s < 4; ++s) {
      const float w = __expf(m[s] - M);   // e^{-inf-M} = 0 for empty segments
      L   += w * l[s];
      acc += w * Obuf[((long)(tile * 4 + s)) * 1024 + e];
    }
    out[(long)(b * TT + i * 16 + row) * HH + (e & 63)] = acc / L;
  }
}

// ---------------------------------------------------------------------------
extern "C" void kernel_launch(void* const* d_in, const int* in_sizes, int n_in,
                              void* d_out, int out_size, void* d_ws,
                              size_t ws_size, hipStream_t stream) {
  (void)in_sizes; (void)n_in; (void)out_size;
  const float* values  = (const float*)d_in[0];
  const float* keys    = (const float*)d_in[1];
  const float* queries = (const float*)d_in[2];
  const float* Wk = (const float*)d_in[3];
  const float* Wq = (const float*)d_in[4];
  const float* Wv = (const float*)d_in[5];

  char* ws = (char*)d_ws;
  _Float16* q16 = (_Float16*)(ws);                        // 2 MB
  _Float16* k16 = (_Float16*)(ws + (2ull << 20));         // 2 MB
  _Float16* vT  = (_Float16*)(ws + (4ull << 20));         // 2 MB
  _Float16* WT  = (_Float16*)(ws + (6ull << 20));         // 384 KB

  wt_build<<<768, 256, 0, stream>>>(Wq, Wk, Wv, WT);
  proj_kernel<<<384, 256, 0, stream>>>(queries, keys, values, WT, q16, k16, vT);

  // ws_size is fixed for the run, so this host-side branch is deterministic.
  if (ws_size >= (25ull << 20)) {
    float* Obuf = (float*)(ws + (8ull << 20));            // 4096 * 4KB = 16 MB
    float* mbuf = (float*)(ws + (24ull << 20));           // 256 KB
    float* lbuf = (float*)(ws + (24ull << 20) + (256u << 10));
    attn_main<4, false><<<1024, 128, 0, stream>>>(q16, k16, vT, Obuf, mbuf, lbuf);
    attn_combine<<<1024, 256, 0, stream>>>(Obuf, mbuf, lbuf, (float*)d_out);
  } else {
    attn_main<1, true><<<256, 128, 0, stream>>>(q16, k16, vT, (float*)d_out,
                                                nullptr, nullptr);
  }
}
